// AFTOptimizedLocal_67379446940303
// MI455X (gfx1250) — matrix operand order repaired and checked by reading, not compiled, not run on hardware
//
#include <hip/hip_runtime.h>

typedef __attribute__((ext_vector_type(16))) __bf16 v16bf;
typedef __attribute__((ext_vector_type(8)))  float  v8f;
typedef __attribute__((ext_vector_type(4)))  unsigned int v4u;
typedef __attribute__((ext_vector_type(8)))  int    v8i;
typedef __attribute__((ext_vector_type(4)))  int    v4i;

#define B_   16
#define T_   2048
#define D_   256
#define WIN_ 128

#ifndef __has_builtin
#define __has_builtin(x) 0
#endif
#if __has_builtin(__builtin_amdgcn_tensor_load_to_lds) && \
    __has_builtin(__builtin_amdgcn_s_wait_tensorcnt)
#define HAVE_TDM 1
#else
#define HAVE_TDM 0
#endif

// ---------------- workspace layout (bytes) ----------------
static constexpr size_t NBTD   = (size_t)B_ * T_ * D_;        // 8388608
static constexpr size_t OFF_QSIG = 0;                          // f32 [B,T,D]  32MB
static constexpr size_t OFF_K    = OFF_QSIG + NBTD * 4;        // f32 [B,T,D]  32MB
static constexpr size_t OFF_V    = OFF_K    + NBTD * 4;        // f32 [B,T,D]  32MB
static constexpr size_t OFF_EKT  = OFF_V    + NBTD * 4;        // bf16 [B,D,T] 16MB
static constexpr size_t OFF_EKVT = OFF_EKT  + NBTD * 2;        // bf16 [B,D,T] 16MB
static constexpr size_t OFF_Y    = OFF_EKVT + NBTD * 2;        // bf16 [B,T,D] 16MB
static constexpr size_t OFF_WT   = OFF_Y    + NBTD * 2;        // bf16 4x[256,256] (transposed)
static constexpr size_t OFF_KMAX = OFF_WT   + (size_t)4 * D_ * D_ * 2;
static constexpr size_t OFF_S0   = OFF_KMAX + (size_t)B_ * D_ * 4;
static constexpr size_t OFF_S1   = OFF_S0   + (size_t)B_ * D_ * 4;
static constexpr size_t OFF_WM   = OFF_S1   + (size_t)B_ * D_ * 4;
static constexpr size_t OFF_E0   = OFF_WM   + (size_t)T_ * 4;

// Max band panel: 9 chunks of 32 -> 288 columns; two panels (expk, ekv), 256 rows bf16.
static constexpr int    MAXWIDTH = 288;
static constexpr size_t SMEM_BYTES = (size_t)2 * D_ * MAXWIDTH * 2;   // 294912 <= 320KB WGP LDS

// ---------------- TDM helper: 2-D tile (global -> LDS), 2-byte elements ----------------
#if HAVE_TDM
__device__ __forceinline__ void tdm_load_2d_b16(unsigned lds_byte_off, const void* gaddr,
                                                unsigned width_elems, unsigned height,
                                                unsigned stride_elems) {
    unsigned long long ga = (unsigned long long)(uintptr_t)gaddr;
    v4u g0;
    g0[0] = 1u;                                           // count=1 (valid), user mode
    g0[1] = lds_byte_off;                                 // lds_addr
    g0[2] = (unsigned)(ga & 0xFFFFFFFFu);                 // global_addr[31:0]
    g0[3] = (unsigned)((ga >> 32) & 0x1FFFFFFu) | (2u << 30);   // global_addr[56:32] | type=2
    v8i g1;
    g1[0] = (int)(1u << 16);                              // data_size=1 (2B), wg_mask=0
    g1[1] = (int)((width_elems & 0xFFFFu) << 16);         // tensor_dim0[15:0]
    g1[2] = (int)(((width_elems >> 16) & 0xFFFFu) |
                  ((height & 0xFFFFu) << 16));            // tensor_dim0[31:16] | tensor_dim1[15:0]
    g1[3] = (int)(((height >> 16) & 0xFFFFu) |
                  ((width_elems & 0xFFFFu) << 16));       // tensor_dim1[31:16] | tile_dim0
    g1[4] = (int)(height & 0xFFFFu);                      // tile_dim1 | tile_dim2=0
    g1[5] = (int)stride_elems;                            // tensor_dim0_stride[31:0]
    g1[6] = 0;                                            // stride hi | dim1_stride lo
    g1[7] = 0;
    v4i z4 = {0, 0, 0, 0};                                // groups 2/3 unused (2-D tensor)
    v8i z8 = {0, 0, 0, 0, 0, 0, 0, 0};                    // extra operand (clang-23 6-arg form)
    __builtin_amdgcn_tensor_load_to_lds(g0, g1, z4, z4, z8, 0);
}
#endif

// ---------------- K1: weights -> transposed bf16 ----------------
__global__ void aft_weights_t(const float* __restrict__ Wq, const float* __restrict__ Wk,
                              const float* __restrict__ Wv, const float* __restrict__ Wo,
                              __bf16* __restrict__ Wt) {
    int idx = blockIdx.x * blockDim.x + threadIdx.x;      // 4*256*256 total
    int w   = idx >> 16;
    int rem = idx & 0xFFFF;
    int n = rem >> 8, k = rem & 255;
    const float* src = (w == 0) ? Wq : (w == 1) ? Wk : (w == 2) ? Wv : Wo;
    Wt[idx] = (__bf16)src[k * D_ + n];
}

// ---------------- K2: fused QKV projection (WMMA bf16) ----------------
__global__ void __launch_bounds__(128)
aft_qkv_gemm(const float* __restrict__ x, const __bf16* __restrict__ Wt,
             float* __restrict__ qsig, float* __restrict__ kout, float* __restrict__ vout) {
    const int lane = threadIdx.x & 31;
    const int wave = threadIdx.x >> 5;
    const int which = blockIdx.y;
    const int t0 = blockIdx.x * 16;
    const __bf16* Wsel = Wt + (size_t)which * D_ * D_;

    v8f acc[4] = {};
    const int m = lane & 15;
    const int hi = (lane & 16) ? 1 : 0;

    for (int kk = 0; kk < D_ / 32; ++kk) {
        const float* abase = x + (size_t)(t0 + m) * D_ + kk * 32 + hi * 16;
        float run[16];
        ((float4*)run)[0] = *(const float4*)(abase + 0);
        ((float4*)run)[1] = *(const float4*)(abase + 4);
        ((float4*)run)[2] = *(const float4*)(abase + 8);
        ((float4*)run)[3] = *(const float4*)(abase + 12);
        v16bf a;
#pragma unroll
        for (int j = 0; j < 16; ++j) a[j] = (__bf16)run[j];

#pragma unroll
        for (int nt = 0; nt < 4; ++nt) {
            int n = wave * 64 + nt * 16 + (lane & 15);
            v16bf bb = *(const v16bf*)(Wsel + (size_t)n * D_ + kk * 32 + hi * 16);
            acc[nt] = __builtin_amdgcn_wmma_f32_16x16x32_bf16(
                false, a, false, bb, (short)0, acc[nt], false, false);
        }
    }

    float* outp = (which == 0) ? qsig : (which == 1) ? kout : vout;
#pragma unroll
    for (int nt = 0; nt < 4; ++nt) {
        int d = wave * 64 + nt * 16 + (lane & 15);
#pragma unroll
        for (int r = 0; r < 8; ++r) {
            int row = t0 + r + hi * 8;
            float val = acc[nt][r];
            if (which == 0)            // sigmoid via v_rcp_f32 (bf16 pipeline downstream)
                val = __builtin_amdgcn_rcpf(1.0f + __expf(-val));
            outp[(size_t)row * D_ + d] = val;
        }
    }
}

// ---------------- K3: per-(b,d) column stats over T ----------------
__global__ void aft_col_stats(const float* __restrict__ kbuf, const float* __restrict__ vbuf,
                              float* __restrict__ kmax, float* __restrict__ S0,
                              float* __restrict__ S1) {
    int tid = blockIdx.x * blockDim.x + threadIdx.x;    // B*D
    int b = tid / D_, d = tid % D_;
    const float* kp = kbuf + (size_t)b * T_ * D_ + d;
    const float* vp = vbuf + (size_t)b * T_ * D_ + d;
    float m = -3.4e38f;
    for (int t = 0; t < T_; ++t) m = fmaxf(m, kp[(size_t)t * D_]);
    float s0 = 0.f, s1 = 0.f;
    for (int t = 0; t < T_; ++t) {
        float e = __expf(kp[(size_t)t * D_] - m);
        s0 += e;
        s1 += e * vp[(size_t)t * D_];
    }
    kmax[tid] = m; S0[tid] = s0; S1[tid] = s1;
}

// ---------------- K4: per-row band stats of w_bias ----------------
__global__ void aft_row_stats(const float* __restrict__ wb, float* __restrict__ wm,
                              float* __restrict__ e0) {
    int t = blockIdx.x * blockDim.x + threadIdx.x;      // T
    int lo = max(0, t - (WIN_ - 1)), hi = min(T_ - 1, t + (WIN_ - 1));
    const float* row = wb + (size_t)t * T_;
    float m = 0.0f;                                     // out-of-band zeros participate
    for (int s = lo; s <= hi; ++s) m = fmaxf(m, row[s]);
    wm[t] = m; e0[t] = __expf(-m);
}

// ---------------- K5: exp + LDS-tiled transpose to bf16 [B,D,T] ----------------
__global__ void __launch_bounds__(256)
aft_expk_transpose(const float* __restrict__ kbuf, const float* __restrict__ vbuf,
                   const float* __restrict__ kmax,
                   __bf16* __restrict__ expkT, __bf16* __restrict__ ekvT) {
    __shared__ __bf16 sk[32][33];
    __shared__ __bf16 sv[32][33];
    const int b = blockIdx.z;
    const int t0 = blockIdx.x * 32;
    const int d0 = blockIdx.y * 32;
    {
        int dl = threadIdx.x & 31, tb = threadIdx.x >> 5;
        float km = kmax[b * D_ + d0 + dl];
#pragma unroll
        for (int i = 0; i < 4; ++i) {
            int tl = tb + i * 8;
            size_t idx = ((size_t)b * T_ + t0 + tl) * D_ + d0 + dl;
            float e = __expf(kbuf[idx] - km);
            sk[tl][dl] = (__bf16)e;
            sv[tl][dl] = (__bf16)(e * vbuf[idx]);
        }
    }
    __syncthreads();
    {
        int tl = threadIdx.x & 31, db = threadIdx.x >> 5;
#pragma unroll
        for (int i = 0; i < 4; ++i) {
            int dl = db + i * 8;
            size_t o = ((size_t)b * D_ + d0 + dl) * T_ + t0 + tl;
            expkT[o] = sk[tl][dl];
            ekvT[o]  = sv[tl][dl];
        }
    }
}

// ---------------- K6: banded AFT core: TDM -> LDS -> WMMA, fused gate ----------------
// num[t,:] = e0_t*S1 + sum_{s in band} (exp(w_ts - m_t) - e0_t) * ekv[s,:]   (same for den)
// grid = (T/16, B), block = 128 (4 waves); wave covers 64 d-columns.
__global__ void __launch_bounds__(128)
aft_band_core(const float* __restrict__ wb, const float* __restrict__ wm,
              const float* __restrict__ e0,
              const __bf16* __restrict__ expkT, const __bf16* __restrict__ ekvT,
              const float* __restrict__ S0, const float* __restrict__ S1,
              const float* __restrict__ qsig, __bf16* __restrict__ ybuf) {
    extern __shared__ __bf16 smem[];                    // [2][256][width] bf16 panels
    const int lane = threadIdx.x & 31;
    const int wave = threadIdx.x >> 5;
    const int b = blockIdx.y;
    const int t0 = blockIdx.x * 16;
    const int hi = (lane & 16) ? 1 : 0;

    const int sb  = max(0, t0 - (WIN_ - 1)) & ~31;      // 32-aligned band start
    const int se  = min(T_, t0 + 15 + WIN_);            // band end (exclusive)
    const int nch = (se - sb + 31) >> 5;                // <= 9 K-chunks of 32
    const int width = nch * 32;                         // panel width in elements

#if HAVE_TDM
    // ---- async tensor DMA of both B panels into LDS (one issue per block) ----
    if (threadIdx.x < 32 && lane == 0) {
        tdm_load_2d_b16(0u, expkT + (size_t)b * D_ * T_ + sb,
                        (unsigned)width, (unsigned)D_, (unsigned)T_);
        tdm_load_2d_b16((unsigned)((size_t)D_ * width * 2),
                        ekvT + (size_t)b * D_ * T_ + sb,
                        (unsigned)width, (unsigned)D_, (unsigned)T_);
    }
#endif

    // ---- overlap DMA: build all correction A-tiles in registers (v_exp co-issues) ----
    const int ta = t0 + (lane & 15);
    const float wmt = wm[ta];
    const float e0t = e0[ta];
    const float* wrow = wb + (size_t)ta * T_;

    v16bf a[9];
    for (int c = 0; c < nch; ++c) {
        const int s_base = sb + c * 32;
#pragma unroll
        for (int j = 0; j < 16; ++j) {
            int kidx = hi * 16 + j;  // contiguous k order, matches B
            int s = s_base + kidx;
            int dlt = ta - s; if (dlt < 0) dlt = -dlt;
            float cv = 0.0f;
            if (dlt < WIN_) cv = __expf(wrow[s] - wmt) - e0t;
            a[c][j] = (__bf16)cv;
        }
    }

#if HAVE_TDM
    if (threadIdx.x < 32) __builtin_amdgcn_s_wait_tensorcnt(0);
    __syncthreads();                                    // LDS panels now visible to all waves
    const __bf16* sEK  = smem;
    const __bf16* sEKV = smem + (size_t)D_ * width;
#endif

    v8f numacc[4] = {};
    v8f denacc[4] = {};

    for (int c = 0; c < nch; ++c) {
        const int s_base = sb + c * 32;
#pragma unroll
        for (int nt = 0; nt < 4; ++nt) {
            int d = wave * 64 + nt * 16 + (lane & 15);
#if HAVE_TDM
            size_t loff = (size_t)d * width + (s_base - sb) + hi * 16;
            v16bf bk = *(const v16bf*)(sEK + loff);      // ds_load_b128 x2
            v16bf bv = *(const v16bf*)(sEKV + loff);
#else
            size_t boff = ((size_t)b * D_ + d) * T_ + s_base + hi * 16;
            v16bf bk = *(const v16bf*)(expkT + boff);
            v16bf bv = *(const v16bf*)(ekvT + boff);
#endif
            denacc[nt] = __builtin_amdgcn_wmma_f32_16x16x32_bf16(
                false, a[c], false, bk, (short)0, denacc[nt], false, false);
            numacc[nt] = __builtin_amdgcn_wmma_f32_16x16x32_bf16(
                false, a[c], false, bv, (short)0, numacc[nt], false, false);
        }
    }

    // ---- epilogue: out-of-band baseline + sigmoid gate (v_rcp_f32, no precise div) ----
    float e0r[8];
#pragma unroll
    for (int r = 0; r < 8; ++r) e0r[r] = e0[t0 + r + hi * 8];

#pragma unroll
    for (int nt = 0; nt < 4; ++nt) {
        int d = wave * 64 + nt * 16 + (lane & 15);
        float s0 = S0[b * D_ + d];
        float s1 = S1[b * D_ + d];
#pragma unroll
        for (int r = 0; r < 8; ++r) {
            int t = t0 + r + hi * 8;
            float den = e0r[r] * s0 + denacc[nt][r];
            float num = e0r[r] * s1 + numacc[nt][r];
            size_t o = ((size_t)b * T_ + t) * D_ + d;
            ybuf[o] = (__bf16)(qsig[o] * num * __builtin_amdgcn_rcpf(den));
        }
    }
}

// ---------------- K7: output projection out = y @ Wo (WMMA bf16) ----------------
__global__ void __launch_bounds__(128)
aft_out_gemm(const __bf16* __restrict__ y, const __bf16* __restrict__ WoT,
             float* __restrict__ out) {
    const int lane = threadIdx.x & 31;
    const int wave = threadIdx.x >> 5;
    const int t0 = blockIdx.x * 16;
    const int hi = (lane & 16) ? 1 : 0;
    const int m = lane & 15;

    v8f acc[4] = {};
    for (int kk = 0; kk < D_ / 32; ++kk) {
        const __bf16* abase = y + (size_t)(t0 + m) * D_ + kk * 32 + hi * 16;
        v16bf a;
        ((int4*)&a)[0] = *(const int4*)(abase + 0);
        ((int4*)&a)[1] = *(const int4*)(abase + 8);
#pragma unroll
        for (int nt = 0; nt < 4; ++nt) {
            int n = wave * 64 + nt * 16 + (lane & 15);
            v16bf bb = *(const v16bf*)(WoT + (size_t)n * D_ + kk * 32 + hi * 16);
            acc[nt] = __builtin_amdgcn_wmma_f32_16x16x32_bf16(
                false, a, false, bb, (short)0, acc[nt], false, false);
        }
    }
#pragma unroll
    for (int nt = 0; nt < 4; ++nt) {
        int d = wave * 64 + nt * 16 + (lane & 15);
#pragma unroll
        for (int r = 0; r < 8; ++r)
            out[(size_t)(t0 + r + hi * 8) * D_ + d] = acc[nt][r];
    }
}

// ---------------- launcher ----------------
extern "C" void kernel_launch(void* const* d_in, const int* in_sizes, int n_in,
                              void* d_out, int out_size, void* d_ws, size_t ws_size,
                              hipStream_t stream) {
    const float* x  = (const float*)d_in[0];
    const float* Wq = (const float*)d_in[1];
    const float* Wk = (const float*)d_in[2];
    const float* Wv = (const float*)d_in[3];
    const float* Wo = (const float*)d_in[4];
    const float* wb = (const float*)d_in[5];
    float* out = (float*)d_out;

    char* ws = (char*)d_ws;
    float*  qsig  = (float*)(ws + OFF_QSIG);
    float*  kbuf  = (float*)(ws + OFF_K);
    float*  vbuf  = (float*)(ws + OFF_V);
    __bf16* expkT = (__bf16*)(ws + OFF_EKT);
    __bf16* ekvT  = (__bf16*)(ws + OFF_EKVT);
    __bf16* ybuf  = (__bf16*)(ws + OFF_Y);
    __bf16* Wt    = (__bf16*)(ws + OFF_WT);
    __bf16* WoT   = Wt + (size_t)3 * D_ * D_;
    float*  kmax  = (float*)(ws + OFF_KMAX);
    float*  S0    = (float*)(ws + OFF_S0);
    float*  S1    = (float*)(ws + OFF_S1);
    float*  wm    = (float*)(ws + OFF_WM);
    float*  e0    = (float*)(ws + OFF_E0);

    aft_weights_t<<<(4 * D_ * D_) / 256, 256, 0, stream>>>(Wq, Wk, Wv, Wo, Wt);
    aft_qkv_gemm<<<dim3((B_ * T_) / 16, 3), 128, 0, stream>>>(x, Wt, qsig, kbuf, vbuf);
    aft_col_stats<<<(B_ * D_) / 256, 256, 0, stream>>>(kbuf, vbuf, kmax, S0, S1);
    aft_row_stats<<<T_ / 256, 256, 0, stream>>>(wb, wm, e0);
    aft_expk_transpose<<<dim3(T_ / 32, D_ / 32, B_), 256, 0, stream>>>(kbuf, vbuf, kmax,
                                                                       expkT, ekvT);
    aft_band_core<<<dim3(T_ / 16, B_), 128, SMEM_BYTES, stream>>>(wb, wm, e0, expkT, ekvT,
                                                                  S0, S1, qsig, ybuf);
    aft_out_gemm<<<(B_ * T_) / 16, 128, 0, stream>>>(ybuf, WoT, out);
}
